// Decoder_31413390803116
// MI455X (gfx1250) — matrix operand order repaired and checked by reading, not compiled, not run on hardware
//
#include <hip/hip_runtime.h>
#include <math.h>

// ---------------- constants ----------------
#define T_STEPS 32
#define BATCH   64
#define SRC     128
#define WV      512
#define ENC     512
#define DEC     512
#define ATT     512
#define IN0     1024   // WV + ENC
#define G3      1536   // 3*DEC
#define RK      1536   // WV + DEC + ENC

typedef __bf16 bf16;
typedef __attribute__((ext_vector_type(16))) __bf16 v16bf;
typedef __attribute__((ext_vector_type(8)))  float  v8f;
typedef __attribute__((ext_vector_type(4)))  unsigned int u32x4;

struct Pack32 { u32x4 a, b; };   // 32 bytes -> one v16bf fragment

__device__ inline float sigm(float x) { return 1.0f / (1.0f + __expf(-x)); }

// ---------------- WMMA GEMM ----------------
// C[M,N] = A[M,K](bf16,row-major) * W[N,K](bf16,row-major)^T + bias[N] (optional)
// grid = (N/64, M/16), block = 128 (4 waves, each wave one 16x16 tile)
__global__ void gemm_wmma_bf16(const bf16* __restrict__ A,
                               const bf16* __restrict__ W,
                               const float* __restrict__ bias,
                               float* __restrict__ C,
                               int M, int N, int K) {
    const int wave = threadIdx.x >> 5;
    const int lane = threadIdx.x & 31;
    const int half = lane >> 4;       // 0: lanes 0-15, 1: lanes 16-31
    const int r    = lane & 15;       // row (A) / col (B) within tile
    const int m0   = blockIdx.y * 16;
    const int n0   = (blockIdx.x * 4 + wave) * 16;

    const u32x4* Arow = reinterpret_cast<const u32x4*>(A + (size_t)(m0 + r) * K);
    const u32x4* Brow = reinterpret_cast<const u32x4*>(W + (size_t)(n0 + r) * K);

    float bv = bias ? bias[n0 + r] : 0.0f;
    v8f acc = {bv, bv, bv, bv, bv, bv, bv, bv};

    for (int k0 = 0; k0 < K; k0 += 32) {
        // A fragment: lane half 0 -> K {k0..k0+7, k0+16..k0+23}
        //             lane half 1 -> K {k0+8..k0+15, k0+24..k0+31}
        int ai = (k0 >> 3) + half;
        Pack32 ap;
        ap.a = Arow[ai];
        ap.b = Arow[ai + 2];
        // B fragment: same K layout as A
        int bi = (k0 >> 3) + half;
        Pack32 bp;
        bp.a = Brow[bi];
        bp.b = Brow[bi + 2];
        v16bf av = __builtin_bit_cast(v16bf, ap);
        v16bf bw = __builtin_bit_cast(v16bf, bp);
        acc = __builtin_amdgcn_wmma_f32_16x16x32_bf16(
            false, av, false, bw, (short)0, acc, false, false);
    }

    // C/D layout: VGPR g, lane l: row = g + 8*(l>>4), col = l&15
    #pragma unroll
    for (int g = 0; g < 8; ++g)
        C[(size_t)(m0 + g + 8 * half) * N + (n0 + r)] = acc[g];
}

// ---------------- small elementwise kernels ----------------
__global__ void f2bf_kernel(const float* __restrict__ in, bf16* __restrict__ out, int n) {
    int i = blockIdx.x * blockDim.x + threadIdx.x;
    if (i < n) out[i] = (bf16)in[i];
}

// embB[row, j] = bf16(emb_W[input[row], j]); rows = T*B
__global__ void gather_emb_kernel(const int* __restrict__ tok,
                                  const float* __restrict__ embW,
                                  bf16* __restrict__ embB, int n) {
    int i = blockIdx.x * blockDim.x + threadIdx.x;
    if (i >= n) return;
    int row = i >> 9, j = i & 511;
    embB[i] = (bf16)embW[(size_t)tok[row] * WV + j];
}

// ctxb[(b*S+s)*ENC+e] = bf16(context[(s*B+b)*ENC+e])
__global__ void ctx_convert_kernel(const float* __restrict__ ctx,
                                   bf16* __restrict__ ctxb, int n) {
    int i = blockIdx.x * blockDim.x + threadIdx.x;
    if (i >= n) return;
    int e = i & 511, rest = i >> 9;
    int s = rest & (SRC - 1), b = rest >> 7;
    ctxb[i] = (bf16)ctx[((size_t)s * BATCH + b) * ENC + e];
}

__global__ void init_state_kernel(const float* __restrict__ hidden,
                                  const float* __restrict__ init_att,
                                  float* __restrict__ h_state,
                                  bf16* __restrict__ Hb0, bf16* __restrict__ Hb1,
                                  float* __restrict__ ctxcur, bf16* __restrict__ ctxcur_bf) {
    int i = blockIdx.x * blockDim.x + threadIdx.x;  // 0..65535
    if (i >= 2 * BATCH * DEC) return;
    float h = hidden[i];
    h_state[i] = h;
    if (i < BATCH * DEC) Hb0[i] = (bf16)h;
    else                 Hb1[i - BATCH * DEC] = (bf16)h;
    if (i < BATCH * ENC) {
        float c = init_att[i];
        ctxcur[i] = c;
        ctxcur_bf[i] = (bf16)c;
    }
}

// X = [emb_t | cur_ctx] (bf16), also seed Rin[:,0:512] with emb_t
__global__ void build_x_kernel(const bf16* __restrict__ embT,
                               const bf16* __restrict__ cbf,
                               bf16* __restrict__ Xb, bf16* __restrict__ Rin) {
    int i = blockIdx.x * blockDim.x + threadIdx.x;  // 0..32767
    if (i >= BATCH * WV) return;
    int b = i >> 9, j = i & 511;
    bf16 e = embT[i];
    Xb[(size_t)b * IN0 + j]       = e;
    Xb[(size_t)b * IN0 + WV + j]  = cbf[i];
    Rin[(size_t)b * RK + j]       = e;
}

// GRU gate fusion: r,z,n from GI/GH [B,3*DEC]; updates h (fp32) + bf16 mirror
__global__ void gru_kernel(const float* __restrict__ GI, const float* __restrict__ GH,
                           float* __restrict__ h, bf16* __restrict__ hb,
                           bf16* __restrict__ extra, int estride, int eoff) {
    int i = blockIdx.x * blockDim.x + threadIdx.x;  // 0..32767
    if (i >= BATCH * DEC) return;
    int b = i >> 9, j = i & 511;
    const float* gi = GI + (size_t)b * G3;
    const float* gh = GH + (size_t)b * G3;
    float r = sigm(gi[j] + gh[j]);
    float z = sigm(gi[DEC + j] + gh[DEC + j]);
    float n = tanhf(gi[2 * DEC + j] + r * gh[2 * DEC + j]);
    float hn = (1.0f - z) * n + z * h[i];
    h[i] = hn;
    hb[i] = (bf16)hn;
    if (extra) extra[(size_t)b * estride + eoff + j] = (bf16)hn;
}

// Fused attention: energy=tanh(pre+q)@w_v, mask, softmax, new_ctx, copy gate
// grid = B blocks, 256 threads
__global__ void attn_kernel(const float* __restrict__ pre,   // [B,S,ATT]
                            const float* __restrict__ Q,     // [B,ATT]
                            const float* __restrict__ wv,    // [ATT]
                            const float* __restrict__ mask,  // [B,S]
                            const float* __restrict__ ctx,   // [S,B,ENC]
                            const float* __restrict__ out_h, // [B,DEC]
                            const float* __restrict__ Wc,    // [ENC+DEC]
                            const float* __restrict__ bc,    // [1]
                            float* __restrict__ attn_out,    // c_outputs + t*B*S
                            float* __restrict__ attn_last,   // or nullptr
                            float* __restrict__ ctxcur,
                            bf16* __restrict__ ctxcur_bf,
                            bf16* __restrict__ Rin,
                            float* __restrict__ copy_out) {  // copy_gates + t*B
    int b = blockIdx.x, tid = threadIdx.x;
    __shared__ float sE[SRC];
    __shared__ float sPart[256];
    __shared__ float sCtx[ENC];
    __shared__ float sRed[256];
    __shared__ float sMx, sSum;

    int s = tid & (SRC - 1), part = tid >> 7;
    const float* prow = pre + ((size_t)b * SRC + s) * ATT;
    const float* q = Q + (size_t)b * ATT;
    float p = 0.0f;
    for (int a = part * 256; a < part * 256 + 256; ++a)
        p += wv[a] * tanhf(prow[a] + q[a]);
    sPart[tid] = p;
    __syncthreads();
    if (tid < SRC) {
        float e = sPart[tid] + sPart[tid + SRC];
        if (mask[(size_t)b * SRC + tid] > 0.5f) e = -3.0e38f;
        sE[tid] = e;
    }
    __syncthreads();
    if (tid == 0) {
        float mx = sE[0];
        for (int k = 1; k < SRC; ++k) mx = fmaxf(mx, sE[k]);
        sMx = mx;
    }
    __syncthreads();
    if (tid < SRC) sE[tid] = __expf(sE[tid] - sMx);
    __syncthreads();
    if (tid == 0) {
        float sum = 0.0f;
        for (int k = 0; k < SRC; ++k) sum += sE[k];
        sSum = sum;
    }
    __syncthreads();
    if (tid < SRC) {
        float av = sE[tid] / sSum;
        sE[tid] = av;
        attn_out[(size_t)b * SRC + tid] = av;
        if (attn_last) attn_last[(size_t)b * SRC + tid] = av;
    }
    __syncthreads();
    // new_ctx[e] = sum_s attn[s] * ctx[s,b,e]
    for (int e = tid; e < ENC; e += 256) {
        float acc = 0.0f;
        for (int k = 0; k < SRC; ++k)
            acc += sE[k] * ctx[((size_t)k * BATCH + b) * ENC + e];
        sCtx[e] = acc;
        ctxcur[(size_t)b * ENC + e] = acc;
        ctxcur_bf[(size_t)b * ENC + e] = (bf16)acc;
        Rin[(size_t)b * RK + WV + DEC + e] = (bf16)acc;
    }
    __syncthreads();
    // copy gate: sigmoid([out, new_ctx] . Wc + bc)
    float cp = 0.0f;
    for (int i = tid; i < DEC + ENC; i += 256) {
        float v = (i < DEC) ? out_h[(size_t)b * DEC + i] : sCtx[i - DEC];
        cp += v * Wc[i];
    }
    sRed[tid] = cp;
    __syncthreads();
    if (tid == 0) {
        float sum = 0.0f;
        for (int k = 0; k < 256; ++k) sum += sRed[k];
        copy_out[b] = sigm(sum + bc[0]);
    }
}

// maxout pool-2 over readout (bias already added in GEMM)
__global__ void maxout_kernel(const float* __restrict__ R, float* __restrict__ gout) {
    int i = blockIdx.x * blockDim.x + threadIdx.x;  // 0..16383
    if (i >= BATCH * (DEC / 2)) return;
    int b = i >> 8, j = i & 255;
    float a = R[(size_t)b * DEC + 2 * j];
    float c = R[(size_t)b * DEC + 2 * j + 1];
    gout[i] = fmaxf(a, c);
}

__global__ void finalize_kernel(const float* __restrict__ h_state,
                                const float* __restrict__ ctxcur,
                                float* __restrict__ out_hid,
                                float* __restrict__ out_ctx) {
    int i = blockIdx.x * blockDim.x + threadIdx.x;
    if (i >= 2 * BATCH * DEC) return;
    out_hid[i] = h_state[i];
    if (i < BATCH * ENC) out_ctx[i] = ctxcur[i];
}

// ---------------- host ----------------
static inline dim3 cdiv256(int n) { return dim3((n + 255) / 256); }

extern "C" void kernel_launch(void* const* d_in, const int* in_sizes, int n_in,
                              void* d_out, int out_size, void* d_ws, size_t ws_size,
                              hipStream_t stream) {
    const int*   tok      = (const int*)d_in[0];
    const float* hidden   = (const float*)d_in[1];
    const float* context  = (const float*)d_in[2];
    const float* mask     = (const float*)d_in[3];
    const float* init_att = (const float*)d_in[4];
    const float* embW     = (const float*)d_in[5];
    const float* W_ih0 = (const float*)d_in[6];  const float* b_ih0 = (const float*)d_in[7];
    const float* W_hh0 = (const float*)d_in[8];  const float* b_hh0 = (const float*)d_in[9];
    const float* W_ih1 = (const float*)d_in[10]; const float* b_ih1 = (const float*)d_in[11];
    const float* W_hh1 = (const float*)d_in[12]; const float* b_hh1 = (const float*)d_in[13];
    const float* W_pre = (const float*)d_in[14]; const float* b_pre = (const float*)d_in[15];
    const float* W_q   = (const float*)d_in[16]; const float* w_v   = (const float*)d_in[17];
    const float* W_cp  = (const float*)d_in[18]; const float* b_cp  = (const float*)d_in[19];
    const float* W_rd  = (const float*)d_in[20]; const float* b_rd  = (const float*)d_in[21];

    float* out = (float*)d_out;
    float* out_g    = out;                                   // [T,B,256]
    float* out_c    = out + T_STEPS * BATCH * 256;           // [T,B,S]
    float* out_copy = out_c + T_STEPS * BATCH * SRC;         // [T,B]
    float* out_hid  = out_copy + T_STEPS * BATCH;            // [2,B,DEC]
    float* out_al   = out_hid + 2 * BATCH * DEC;             // [B,S]
    float* out_ctx  = out_al + BATCH * SRC;                  // [B,ENC]

    // workspace carve-up
    char* w = (char*)d_ws;
    auto alloc = [&](size_t bytes) -> char* {
        char* p = w; w += (bytes + 255) & ~(size_t)255; return p;
    };
    bf16* Wih0b  = (bf16*)alloc((size_t)G3 * IN0 * 2);
    bf16* Whh0b  = (bf16*)alloc((size_t)G3 * DEC * 2);
    bf16* Wih1b  = (bf16*)alloc((size_t)G3 * DEC * 2);
    bf16* Whh1b  = (bf16*)alloc((size_t)G3 * DEC * 2);
    bf16* Wqb    = (bf16*)alloc((size_t)ATT * DEC * 2);
    bf16* Wrdb   = (bf16*)alloc((size_t)DEC * RK * 2);
    bf16* Wpreb  = (bf16*)alloc((size_t)ATT * ENC * 2);
    bf16* embB   = (bf16*)alloc((size_t)T_STEPS * BATCH * WV * 2);
    bf16* ctxb   = (bf16*)alloc((size_t)BATCH * SRC * ENC * 2);
    float* pre   = (float*)alloc((size_t)BATCH * SRC * ATT * 4);
    float* h_st  = (float*)alloc((size_t)2 * BATCH * DEC * 4);
    bf16* Hb0    = (bf16*)alloc((size_t)BATCH * DEC * 2);
    bf16* Hb1    = (bf16*)alloc((size_t)BATCH * DEC * 2);
    float* ctxc  = (float*)alloc((size_t)BATCH * ENC * 4);
    bf16* ctxcb  = (bf16*)alloc((size_t)BATCH * ENC * 2);
    bf16* Xb     = (bf16*)alloc((size_t)BATCH * IN0 * 2);
    bf16* Rin    = (bf16*)alloc((size_t)BATCH * RK * 2);
    float* GI0   = (float*)alloc((size_t)BATCH * G3 * 4);
    float* GH0   = (float*)alloc((size_t)BATCH * G3 * 4);
    float* GI1   = (float*)alloc((size_t)BATCH * G3 * 4);
    float* GH1   = (float*)alloc((size_t)BATCH * G3 * 4);
    float* Qb    = (float*)alloc((size_t)BATCH * ATT * 4);
    float* Rb    = (float*)alloc((size_t)BATCH * DEC * 4);

    // ---- one-time prep (re-run every call: deterministic) ----
    f2bf_kernel<<<cdiv256(G3 * IN0), 256, 0, stream>>>(W_ih0, Wih0b, G3 * IN0);
    f2bf_kernel<<<cdiv256(G3 * DEC), 256, 0, stream>>>(W_hh0, Whh0b, G3 * DEC);
    f2bf_kernel<<<cdiv256(G3 * DEC), 256, 0, stream>>>(W_ih1, Wih1b, G3 * DEC);
    f2bf_kernel<<<cdiv256(G3 * DEC), 256, 0, stream>>>(W_hh1, Whh1b, G3 * DEC);
    f2bf_kernel<<<cdiv256(ATT * DEC), 256, 0, stream>>>(W_q, Wqb, ATT * DEC);
    f2bf_kernel<<<cdiv256(DEC * RK), 256, 0, stream>>>(W_rd, Wrdb, DEC * RK);
    f2bf_kernel<<<cdiv256(ATT * ENC), 256, 0, stream>>>(W_pre, Wpreb, ATT * ENC);
    gather_emb_kernel<<<cdiv256(T_STEPS * BATCH * WV), 256, 0, stream>>>(
        tok, embW, embB, T_STEPS * BATCH * WV);
    ctx_convert_kernel<<<cdiv256(BATCH * SRC * ENC), 256, 0, stream>>>(
        context, ctxb, BATCH * SRC * ENC);
    init_state_kernel<<<cdiv256(2 * BATCH * DEC), 256, 0, stream>>>(
        hidden, init_att, h_st, Hb0, Hb1, ctxc, ctxcb);
    // precompute[b,s,:] = ctx_bse @ W_pre^T + b_pre   (M=8192, N=512, K=512)
    gemm_wmma_bf16<<<dim3(ATT / 64, (BATCH * SRC) / 16), 128, 0, stream>>>(
        ctxb, Wpreb, b_pre, pre, BATCH * SRC, ATT, ENC);

    // ---- sequential decode loop ----
    for (int t = 0; t < T_STEPS; ++t) {
        build_x_kernel<<<cdiv256(BATCH * WV), 256, 0, stream>>>(
            embB + (size_t)t * BATCH * WV, ctxcb, Xb, Rin);
        // layer-0 GRU
        gemm_wmma_bf16<<<dim3(G3 / 64, BATCH / 16), 128, 0, stream>>>(
            Xb, Wih0b, b_ih0, GI0, BATCH, G3, IN0);
        gemm_wmma_bf16<<<dim3(G3 / 64, BATCH / 16), 128, 0, stream>>>(
            Hb0, Whh0b, b_hh0, GH0, BATCH, G3, DEC);
        gru_kernel<<<cdiv256(BATCH * DEC), 256, 0, stream>>>(
            GI0, GH0, h_st, Hb0, (bf16*)nullptr, 0, 0);
        // layer-1 GRU (input = h0_new = Hb0)
        gemm_wmma_bf16<<<dim3(G3 / 64, BATCH / 16), 128, 0, stream>>>(
            Hb0, Wih1b, b_ih1, GI1, BATCH, G3, DEC);
        gemm_wmma_bf16<<<dim3(G3 / 64, BATCH / 16), 128, 0, stream>>>(
            Hb1, Whh1b, b_hh1, GH1, BATCH, G3, DEC);
        gru_kernel<<<cdiv256(BATCH * DEC), 256, 0, stream>>>(
            GI1, GH1, h_st + BATCH * DEC, Hb1, Rin, RK, WV);
        // attention query q = out @ W_q^T
        gemm_wmma_bf16<<<dim3(ATT / 64, BATCH / 16), 128, 0, stream>>>(
            Hb1, Wqb, (const float*)nullptr, Qb, BATCH, ATT, DEC);
        // fused energy/softmax/context/copy-gate
        attn_kernel<<<BATCH, 256, 0, stream>>>(
            pre, Qb, w_v, mask, context, h_st + BATCH * DEC, W_cp, b_cp,
            out_c + (size_t)t * BATCH * SRC,
            (t == T_STEPS - 1) ? out_al : (float*)nullptr,
            ctxc, ctxcb, Rin, out_copy + (size_t)t * BATCH);
        // readout + maxout
        gemm_wmma_bf16<<<dim3(DEC / 64, BATCH / 16), 128, 0, stream>>>(
            Rin, Wrdb, b_rd, Rb, BATCH, DEC, RK);
        maxout_kernel<<<cdiv256(BATCH * DEC / 2), 256, 0, stream>>>(
            Rb, out_g + (size_t)t * BATCH * (DEC / 2));
    }

    finalize_kernel<<<cdiv256(2 * BATCH * DEC), 256, 0, stream>>>(
        h_st, ctxc, out_hid, out_ctx);
    (void)in_sizes; (void)n_in; (void)out_size; (void)ws_size;
}
